// AttnResModule2D_30313879175635
// MI455X (gfx1250) — matrix operand order repaired and checked by reading, not compiled, not run on hardware
//
#include <hip/hip_runtime.h>
#include <hip/hip_bf16.h>

typedef __attribute__((ext_vector_type(16))) __bf16 v16bf;
typedef __attribute__((ext_vector_type(4)))  __bf16 v4bf;
typedef __attribute__((ext_vector_type(8)))  float  v8f;

namespace {
constexpr int cL  = 9;     // 8 history layers + current
constexpr int cB  = 2;
constexpr int cS  = 2048;
constexpr int cD  = 1024;
constexpr int cH  = 16;
constexpr int cDH = 64;    // head dim
constexpr float cLN_EPS = 1e-5f;
constexpr float cQSCALE = 0.125f; // 1/sqrt(64), folded into Q
}

// ---------------- WMMA helpers (CDNA5 wave32 layouts) ----------------

__device__ inline v8f wmma_bf16(v16bf a, v16bf b, v8f c) {
  return __builtin_amdgcn_wmma_f32_16x16x32_bf16(
      /*neg_a=*/false, a, /*neg_b=*/false, b,
      /*c_mod=*/(short)0, c, /*reuse_a=*/false, /*reuse_b=*/false);
}

// A-matrix 16x32 bf16, row-major source base[m*ld + k].
// lanes 0-15: V0..7 -> K=0..15;
// lanes16-31: V0..7 -> K=16..31; M = lane%16.
__device__ inline v16bf load_a_frag(const __bf16* base, int ld) {
  const int lane = threadIdx.x & 31;
  const int m = lane & 15, half = lane >> 4;
  const __bf16* p = base + (size_t)m * ld;
  v16bf a;
#pragma unroll
  for (int v = 0; v < 8; ++v) {
    const int kk = 2 * v + 16 * half;
    a[2 * v]     = p[kk];
    a[2 * v + 1] = p[kk + 1];
  }
  return a;
}

// B fragment from swizzled storage: each lane's 16 elements are contiguous
// (32B, 32B-aligned) -> 2x global_load_b128, coalesced across the wave.
__device__ inline v16bf load_b_sw(const __bf16* fragbase) {
  const int lane = threadIdx.x & 31;
  return *(const v16bf*)(fragbase + (size_t)lane * 16);
}

// Reductions over each 16-lane half (xor of bits 0..3 never crosses halves).
__device__ inline float hmax16(float x) {
#pragma unroll
  for (int m = 1; m < 16; m <<= 1) x = fmaxf(x, __shfl_xor(x, m, 32));
  return x;
}
__device__ inline float hsum16(float x) {
#pragma unroll
  for (int m = 1; m < 16; m <<= 1) x += __shfl_xor(x, m, 32);
  return x;
}

// ---------------- Kernel 1: fused LN + bf16 K/V + fragment swizzles ----
// One 256-thread block per (l,b,s) row of V_col = concat(history, current).
// For the last layer (== current) it additionally emits the row-attention
// K/V in WMMA-B-fragment-major layout (Ksw / Vsw).
__global__ __launch_bounds__(256)
void ln_prep_kernel(const float* __restrict__ hist, const float* __restrict__ cur,
                    const float* __restrict__ gamma, const float* __restrict__ beta,
                    __bf16* __restrict__ Kbf, __bf16* __restrict__ Vbf,
                    __bf16* __restrict__ Ksw, __bf16* __restrict__ Vsw) {
  const int row = blockIdx.x;            // (l*B + b)*S + s
  const int l   = row / (cB * cS);
  const int bs  = row % (cB * cS);
  const float* src = (l < cL - 1) ? (hist + (size_t)row * cD)
                                  : (cur + (size_t)bs * cD);
  const int tid = threadIdx.x;
  const float4 x = ((const float4*)src)[tid];
  float s  = x.x + x.y + x.z + x.w;
  float s2 = x.x * x.x + x.y * x.y + x.z * x.z + x.w * x.w;
#pragma unroll
  for (int m = 1; m < 32; m <<= 1) { s += __shfl_xor(s, m, 32); s2 += __shfl_xor(s2, m, 32); }

  __shared__ float rs[8], rq[8];
  __shared__ float mu_s, rstd_s;
  const int wid = tid >> 5;
  if ((tid & 31) == 0) { rs[wid] = s; rq[wid] = s2; }
  __syncthreads();
  if (tid == 0) {
    float ts = 0.f, tq = 0.f;
#pragma unroll
    for (int i = 0; i < 8; ++i) { ts += rs[i]; tq += rq[i]; }
    const float mu  = ts / cD;
    const float var = tq / cD - mu * mu;
    mu_s = mu;  rstd_s = rsqrtf(var + cLN_EPS);
  }
  __syncthreads();
  const float mu = mu_s, rstd = rstd_s;
  const float4 g  = ((const float4*)gamma)[tid];
  const float4 be = ((const float4*)beta)[tid];

  __bf16 kb[4], vb[4];
  kb[0] = (__bf16)((x.x - mu) * rstd * g.x + be.x);
  kb[1] = (__bf16)((x.y - mu) * rstd * g.y + be.y);
  kb[2] = (__bf16)((x.z - mu) * rstd * g.z + be.z);
  kb[3] = (__bf16)((x.w - mu) * rstd * g.w + be.w);
  vb[0] = (__bf16)x.x; vb[1] = (__bf16)x.y; vb[2] = (__bf16)x.z; vb[3] = (__bf16)x.w;

  __bf16* kp = Kbf + (size_t)row * cD + tid * 4;
  __bf16* vp = Vbf + (size_t)row * cD + tid * 4;
  *(uint2*)kp = *(const uint2*)kb;
  *(uint2*)vp = *(const uint2*)vb;

  if (l == cL - 1) {           // row-attention operands, fragment-major
    const int b = bs / cS, skey = bs % cS;
    const int d0 = tid * 4;
    {  // K score-fragment: B[k=dim][n=key]; lane = (dim16half)*16 + key%16, e = dim%16
      const int h = d0 >> 6, kc = (d0 >> 5) & 1, kh = (d0 >> 4) & 1, e = d0 & 15;
      const int st = skey >> 4, n = skey & 15;
      const int lane = kh * 16 + n;
      __bf16* dst = Ksw + ((((((size_t)b * cH + h) * (cS / 16) + st) * 2 + kc) * 32 + lane) * 16 + e);
      *(uint2*)dst = *(const uint2*)kb;   // contiguous 4 elements (8B)
    }
    {  // V PV-fragment: B[k=key][n=dim]; lane = (key16half)*16 + dim%16, e = key%16
      const int h = d0 >> 6, kt = skey >> 5, kh = (skey >> 4) & 1, e = skey & 15;
#pragma unroll
      for (int j = 0; j < 4; ++j) {
        const int d = d0 + j;
        const int c = (d >> 4) & 3, n = d & 15;
        const int lane = kh * 16 + n;
        Vsw[(((((size_t)b * cH + h) * (cS / 32) + kt) * 4 + c) * 32 + lane) * 16 + e] = vb[j];
      }
    }
  }
}

// ---------------- Kernel 2: Wq -> fragment-major bf16 (WqS) ----------
// WqS[kc][tj][lane][e] = Wq[tj*16 + lane%16][kc*32 + (lane/16)*16 + e]
__global__ __launch_bounds__(256)
void wqs_kernel(const float* __restrict__ Wq, __bf16* __restrict__ WqS) {
  const int t = blockIdx.x * 256 + threadIdx.x;   // 262144 threads, 4 elems each
  const int e4   = t & 3;
  const int lane = (t >> 2) & 31;
  const int tj   = (t >> 7) & 63;
  const int kc   = t >> 13;
  const int j = tj * 16 + (lane & 15);
  const int k = kc * 32 + (lane >> 4) * 16 + e4 * 4;
  const float4 w = *(const float4*)(Wq + (size_t)j * cD + k);
  __bf16* dst = WqS + ((((size_t)kc * (cD / 16) + tj) * 32 + lane) * 16 + e4 * 4);
  dst[0] = (__bf16)w.x; dst[1] = (__bf16)w.y; dst[2] = (__bf16)w.z; dst[3] = (__bf16)w.w;
}

// ---------------- Kernel 3: Q = (current @ Wq^T) * 1/sqrt(d) ----------
// One wave computes a 16x64 strip: A fragment reused across 4 column tiles.
__global__ __launch_bounds__(32)
void qgemm_kernel(const __bf16* __restrict__ curbf, const __bf16* __restrict__ WqS,
                  __bf16* __restrict__ Qbf) {
  const int blk = blockIdx.x;
  const int ti  = blk >> 4;                 // 256 row tiles
  const int tj0 = (blk & 15) * 4;           // 4 column tiles per wave
  const int lane = threadIdx.x & 31;
  const __bf16* abase = curbf + (size_t)(ti * 16) * cD;
  v8f acc[4] = {};
#pragma unroll 4
  for (int k0 = 0; k0 < cD; k0 += 32) {
    const v16bf a = load_a_frag(abase + k0, cD);
    const size_t kc = (size_t)(k0 >> 5);
#pragma unroll
    for (int jj = 0; jj < 4; ++jj) {
      const v16bf b = load_b_sw(WqS + ((kc * (cD / 16) + tj0 + jj) * 32) * 16);
      acc[jj] = wmma_bf16(a, b, acc[jj]);
    }
  }
  const int n = lane & 15, half = lane >> 4;
#pragma unroll
  for (int jj = 0; jj < 4; ++jj)
#pragma unroll
    for (int r = 0; r < 8; ++r)
      Qbf[(size_t)(ti * 16 + r + 8 * half) * cD + (tj0 + jj) * 16 + n] =
          (__bf16)(acc[jj][r] * cQSCALE);
}

// ---------------- Kernel 4: flash attention (col keys + causal rows) --
__global__ __launch_bounds__(32)
void attn_kernel(const __bf16* __restrict__ Qbf, const __bf16* __restrict__ Kbf,
                 const __bf16* __restrict__ Vbf, const __bf16* __restrict__ Ksw,
                 const __bf16* __restrict__ Vsw, float* __restrict__ out) {
  const int qt = blockIdx.x, h = blockIdx.y, bb = blockIdx.z;
  const int lane = threadIdx.x & 31;
  const int n = lane & 15, half = lane >> 4;
  const int qrow0 = qt * 16;

  __shared__ __align__(16) __bf16 lds_p[16 * 32];  // P tile (queries x 32 keys)

  const __bf16* qbase = Qbf + ((size_t)bb * cS + qrow0) * cD + h * cDH;
  const v16bf aq0 = load_a_frag(qbase, cD);        // dims 0..31
  const v16bf aq1 = load_a_frag(qbase + 32, cD);   // dims 32..63

  v8f o[4];                 // o[c][r] = out[q=r+8*half][c*16+n]
  float mrow[8], lrow[8];   // running max / sum, shared within each 16-lane half

  // ---- column part: 9 query-dependent layer keys, pure VALU ----
#pragma unroll
  for (int r = 0; r < 8; ++r) {
    const int q = qrow0 + r + 8 * half;
    const __bf16* qp = Qbf + ((size_t)bb * cS + q) * cD + h * cDH;
    const v4bf q4 = *(const v4bf*)(qp + 4 * n);
    float sc[cL];
#pragma unroll
    for (int l = 0; l < cL; ++l) {
      const __bf16* kp = Kbf + (((size_t)l * cB + bb) * cS + q) * cD + h * cDH;
      const v4bf k4 = *(const v4bf*)(kp + 4 * n);
      float p = 0.f;
#pragma unroll
      for (int j = 0; j < 4; ++j) p += (float)q4[j] * (float)k4[j];
      sc[l] = hsum16(p);    // dot over 64 dims (4/lane x 16 lanes)
    }
    float mx = sc[0];
#pragma unroll
    for (int l = 1; l < cL; ++l) mx = fmaxf(mx, sc[l]);
    float pl[cL], ssum = 0.f;
#pragma unroll
    for (int l = 0; l < cL; ++l) { pl[l] = __expf(sc[l] - mx); ssum += pl[l]; }
    mrow[r] = mx; lrow[r] = ssum;
#pragma unroll
    for (int c = 0; c < 4; ++c) {
      float acc = 0.f;
#pragma unroll
      for (int l = 0; l < cL; ++l)
        acc += pl[l] * (float)Vbf[(((size_t)l * cB + bb) * cS + q) * cD
                                  + h * cDH + c * 16 + n];
      o[c][r] = acc;
    }
  }

  // ---- row part: causal flash over 32 keys per step, all-WMMA ----
  const __bf16* kswb = Ksw + ((size_t)bb * cH + h) * (cS / 16) * 1024; // 2*512 elems per 16-key tile
  const __bf16* vswb = Vsw + ((size_t)bb * cH + h) * (cS / 32) * 2048; // 4*512 elems per 32-key tile
  const int nsteps = (qt >> 1) + 1;

  for (int step = 0; step < nsteps; ++step) {
    const int key0 = step * 32;
    const int stK  = step * 2;        // first 16-key score tile index

    if (step + 1 < nsteps) {
      __builtin_prefetch(kswb + (size_t)(stK + 2) * 1024, 0, 1);
      __builtin_prefetch(vswb + (size_t)(step + 1) * 2048, 0, 1);
    }

    // scores: two 16-key tiles x two 32-dim chunks -> 4 WMMAs
    v8f s0 = {}, s1 = {};
    {
      const __bf16* f0 = kswb + (size_t)stK * 1024;
      s0 = wmma_bf16(aq0, load_b_sw(f0), s0);
      s0 = wmma_bf16(aq1, load_b_sw(f0 + 512), s0);
      const __bf16* f1 = f0 + 1024;
      s1 = wmma_bf16(aq0, load_b_sw(f1), s1);
      s1 = wmma_bf16(aq1, load_b_sw(f1 + 512), s1);
    }

    // causal mask (only near the diagonal)
    if (key0 + 31 >= qrow0) {
#pragma unroll
      for (int r = 0; r < 8; ++r) {
        const int q = qrow0 + r + 8 * half;
        s0[r] = (key0 + n      > q) ? -1e30f : s0[r];
        s1[r] = (key0 + 16 + n > q) ? -1e30f : s1[r];
      }
    }

    // online softmax; write P (bf16) to LDS in [q][key] order for the A-frag
#pragma unroll
    for (int r = 0; r < 8; ++r) {
      const float a0 = s0[r], a1 = s1[r];
      const float mnew = fmaxf(mrow[r], hmax16(fmaxf(a0, a1)));
      const float corr = __expf(mrow[r] - mnew);
      const float p0 = __expf(a0 - mnew);
      const float p1 = __expf(a1 - mnew);
      lrow[r] = lrow[r] * corr + hsum16(p0 + p1);
      mrow[r] = mnew;
      o[0][r] *= corr; o[1][r] *= corr; o[2][r] *= corr; o[3][r] *= corr;
      lds_p[(r + 8 * half) * 32 + n]      = (__bf16)p0;
      lds_p[(r + 8 * half) * 32 + 16 + n] = (__bf16)p1;
    }
    __syncthreads();

    // O += P(16x32) @ V(32x64): 4 WMMAs, V fragments straight from L2
    const v16bf ap = load_a_frag(lds_p, 32);
    const __bf16* vf = vswb + (size_t)step * 2048;
#pragma unroll
    for (int c = 0; c < 4; ++c)
      o[c] = wmma_bf16(ap, load_b_sw(vf + (size_t)c * 512), o[c]);
    __syncthreads();
  }

  // epilogue: normalize and store fp32
#pragma unroll
  for (int r = 0; r < 8; ++r) {
    const float inv = 1.f / lrow[r];
    const int q = qrow0 + r + 8 * half;
    float* op = out + ((size_t)bb * cS + q) * cD + h * cDH;
    op[0 * 16 + n] = o[0][r] * inv;
    op[1 * 16 + n] = o[1][r] * inv;
    op[2 * 16 + n] = o[2][r] * inv;
    op[3 * 16 + n] = o[3][r] * inv;
  }
}

// ---------------- launch ----------------
extern "C" void kernel_launch(void* const* d_in, const int* in_sizes, int n_in,
                              void* d_out, int out_size, void* d_ws, size_t ws_size,
                              hipStream_t stream) {
  const float* hist = (const float*)d_in[0];   // (8,B,S,D)
  const float* cur  = (const float*)d_in[1];   // (B,S,D)
  const float* Wq   = (const float*)d_in[2];   // (D,D)
  const float* g    = (const float*)d_in[3];   // (D)
  const float* be   = (const float*)d_in[4];   // (D)
  float* out = (float*)d_out;                  // (B,S,D) fp32

  // workspace partition (~178 MB total); all chunks are >=4KB aligned
  char* ws = (char*)d_ws;
  const size_t kvBytes = (size_t)cL * cB * cS * cD * sizeof(__bf16);  // 75.5 MB
  const size_t qBytes  = (size_t)cB * cS * cD * sizeof(__bf16);       //  8.4 MB
  const size_t wBytes  = (size_t)cD * cD * sizeof(__bf16);            //  2.0 MB
  size_t off = 0;
  __bf16* Kbf = (__bf16*)(ws + off); off += kvBytes;   // LN(V_col), row-major
  __bf16* Vbf = (__bf16*)(ws + off); off += kvBytes;   // V_col, row-major
  __bf16* Qbf = (__bf16*)(ws + off); off += qBytes;    // scaled Q, row-major
  __bf16* WqS = (__bf16*)(ws + off); off += wBytes;    // Wq, fragment-major
  __bf16* Ksw = (__bf16*)(ws + off); off += qBytes;    // K_row, fragment-major
  __bf16* Vsw = (__bf16*)(ws + off); off += qBytes;    // V_row, fragment-major

  ln_prep_kernel<<<cL * cB * cS, 256, 0, stream>>>(hist, cur, g, be,
                                                   Kbf, Vbf, Ksw, Vsw);
  wqs_kernel<<<(cD * cD / 4) / 256, 256, 0, stream>>>(Wq, WqS);

  const __bf16* curbf = Vbf + (size_t)(cL - 1) * cB * cS * cD;
  qgemm_kernel<<<(cB * cS / 16) * (cD / 64), 32, 0, stream>>>(curbf, WqS, Qbf);

  dim3 agrid(cS / 16, cH, cB);
  attn_kernel<<<agrid, 32, 0, stream>>>(Qbf, Kbf, Vbf, Ksw, Vsw, out);
}
